// Parallel_GCN_83562883711802
// MI455X (gfx1250) — matrix operand order repaired and checked by reading, not compiled, not run on hardware
//
#include <hip/hip_runtime.h>
#include <hip/hip_bf16.h>

typedef __attribute__((ext_vector_type(16))) _Float16 v16h;
typedef __attribute__((ext_vector_type(8)))  float    v8f;

#define DIM 256

// ---------------- utility kernels ----------------

__global__ void k_zero_f4(float4* __restrict__ p, long n4) {
    long i = (long)blockIdx.x * blockDim.x + threadIdx.x;
    if (i < n4) p[i] = make_float4(0.f, 0.f, 0.f, 0.f);
}

__global__ void k_zero_f(float* __restrict__ p, int n) {
    int i = blockIdx.x * blockDim.x + threadIdx.x;
    if (i < n) p[i] = 0.f;
}

// degree count via atomics on src
__global__ void k_degree(const int* __restrict__ src, float* __restrict__ deg, int E) {
    int i = blockIdx.x * blockDim.x + threadIdx.x;
    if (i < E) atomicAdd(&deg[src[i]], 1.0f);
}

// norm = clip(deg,1,inf)^-0.5, in place
__global__ void k_norm(float* __restrict__ deg, int n) {
    int i = blockIdx.x * blockDim.x + threadIdx.x;
    if (i < n) {
        float d = deg[i];
        d = d < 1.0f ? 1.0f : d;
        deg[i] = rsqrtf(d);
    }
}

// h = feature ; acc = feature   (float4 granularity, n4 = N*64)
__global__ void k_init(const float4* __restrict__ f, float4* __restrict__ h,
                       float4* __restrict__ acc, long n4) {
    long i = (long)blockIdx.x * blockDim.x + threadIdx.x;
    if (i < n4) { float4 v = f[i]; h[i] = v; acc[i] = v; }
}

// hn[dst] += h[src] * norm[src] ; one thread per (edge, 4-float chunk)
__global__ void k_scatter(const float* __restrict__ h, const float* __restrict__ norm,
                          const int* __restrict__ src, const int* __restrict__ dst,
                          float* __restrict__ hn, long total) {
    long idx = (long)blockIdx.x * blockDim.x + threadIdx.x;
    if (idx >= total) return;
    int e = (int)(idx >> 6);          // 64 chunks of float4 per edge
    int c = ((int)idx & 63) << 2;
    int s = src[e], d = dst[e];
    float ns = norm[s];
    const float4 v = *(const float4*)(h + (long)s * DIM + c);
    float* p = hn + (long)d * DIM + c;
    atomicAdd(p + 0, v.x * ns);
    atomicAdd(p + 1, v.y * ns);
    atomicAdd(p + 2, v.z * ns);
    atomicAdd(p + 3, v.w * ns);
}

// h = hn * norm[row] ; acc += h   (float4 granularity, n4 = N*64)
__global__ void k_post(float* __restrict__ h, const float* __restrict__ hn,
                       const float* __restrict__ norm, float* __restrict__ acc, long n4) {
    long i = (long)blockIdx.x * blockDim.x + threadIdx.x;
    if (i >= n4) return;
    int node = (int)(i >> 6);
    float nv = norm[node];
    float4 v = ((const float4*)hn)[i];
    v.x *= nv; v.y *= nv; v.z *= nv; v.w *= nv;
    ((float4*)h)[i] = v;
    float4 a = ((float4*)acc)[i];
    a.x += v.x; a.y += v.y; a.z += v.z; a.w += v.w;
    ((float4*)acc)[i] = a;
}

// ---------------- WMMA GEMM: out = relu(acc @ W + 4*b) ----------------
// Each wave owns one 16-col N-tile, holds the full K=256 strip of B (f16) in
// 8 register fragments, and streams 16-row M-tiles. f32 accumulation.
__global__ __launch_bounds__(256) void k_gemm_wmma(
        const float* __restrict__ acc, const float* __restrict__ Wm,
        const float* __restrict__ bias, float* __restrict__ out, int M) {
    const int lane  = threadIdx.x & 31;
    const int wave  = (blockIdx.x * (blockDim.x >> 5)) + (threadIdx.x >> 5);
    const int nWavesTotal = gridDim.x * (blockDim.x >> 5);
    const int ntile = wave & 15;               // 256 cols / 16 = 16 tiles
    const int mWave = wave >> 4;
    const int mStride = nWavesTotal >> 4;

    const int col = ntile * 16 + (lane & 15);
    const int hi  = lane >> 4;                 // which half-wave

    // B fragments: 32x16 f16, lanes 0-15 hold K=0..15, lanes 16-31 K=16..31
    v16h bfrag[8];
#pragma unroll
    for (int s = 0; s < 8; ++s) {
        const int kb = s * 32 + hi * 16;
#pragma unroll
        for (int j = 0; j < 16; ++j)
            bfrag[s][j] = (_Float16)Wm[(kb + j) * DIM + col];
    }
    const float bv = 4.0f * bias[col];

    const int rowInTile = lane & 15;
    const int koA = hi * 16;                   // A: lanes 0-15 K 0-15, lanes 16-31 K 16-31
    const int mTiles = M >> 4;                 // 50000/16 = 3125 exact

    for (int mt = mWave; mt < mTiles; mt += mStride) {
        const float* arow = acc + (long)(mt * 16 + rowInTile) * DIM;
        v8f c = {};
#pragma unroll
        for (int s = 0; s < 8; ++s) {
            const int k0 = s * 32 + koA;
            v16h a;
#pragma unroll
            for (int q = 0; q < 8; ++q) a[q]     = (_Float16)arow[k0 + q];
#pragma unroll
            for (int q = 0; q < 8; ++q) a[8 + q] = (_Float16)arow[k0 + 8 + q];
            c = __builtin_amdgcn_wmma_f32_16x16x32_f16(
                    false, a, false, bfrag[s], (short)0, c, false, false);
        }
        // C/D layout: VGPR r -> row mt*16 + r + hi*8, col = lane&15 of tile
        const long rbase = (long)(mt * 16 + hi * 8);
#pragma unroll
        for (int r = 0; r < 8; ++r) {
            float v = c[r] + bv;
            out[(rbase + r) * DIM + col] = v > 0.f ? v : 0.f;
        }
    }
}

// ---------------- launch ----------------

extern "C" void kernel_launch(void* const* d_in, const int* in_sizes, int n_in,
                              void* d_out, int out_size, void* d_ws, size_t ws_size,
                              hipStream_t stream) {
    const float* feature = (const float*)d_in[0];
    const float* Wm      = (const float*)d_in[1];
    const float* bias    = (const float*)d_in[2];
    const int*   src     = (const int*)d_in[3];
    const int*   dst     = (const int*)d_in[4];
    float*       out     = (float*)d_out;

    const int N = in_sizes[0] / DIM;   // 50000
    const int E = in_sizes[3];         // 800000

    float* ws   = (float*)d_ws;
    const size_t normPad = ((size_t)N + 255) & ~(size_t)255;
    float* norm = ws;                                  // N (deg then norm)
    float* h    = ws + normPad;                        // N*256
    float* hn   = h  + (size_t)N * DIM;                // N*256
    float* acc  = hn + (size_t)N * DIM;                // N*256

    const long n4     = (long)N * (DIM / 4);           // float4 count per matrix
    const int  blkN4  = (int)((n4 + 255) / 256);
    const long scTot  = (long)E * (DIM / 4);           // scatter threads
    const int  blkSc  = (int)((scTot + 255) / 256);

    // degree -> norm
    k_zero_f<<<(N + 255) / 256, 256, 0, stream>>>(norm, N);
    k_degree<<<(E + 255) / 256, 256, 0, stream>>>(src, norm, E);
    k_norm<<<(N + 255) / 256, 256, 0, stream>>>(norm, N);

    // h = acc = feature
    k_init<<<blkN4, 256, 0, stream>>>((const float4*)feature, (float4*)h, (float4*)acc, n4);

    // 3 propagation hops (LAMBDA=0 -> unit weights, accumulate into acc)
    for (int hop = 0; hop < 3; ++hop) {
        k_zero_f4<<<blkN4, 256, 0, stream>>>((float4*)hn, n4);
        k_scatter<<<blkSc, 256, 0, stream>>>(h, norm, src, dst, hn, scTot);
        k_post<<<blkN4, 256, 0, stream>>>(h, hn, norm, acc, n4);
    }

    // single fused GEMM: out = relu(acc @ W + 4*b)
    k_gemm_wmma<<<512, 256, 0, stream>>>(acc, Wm, bias, out, N);
}
